// AttentionAggregator3_54494545052270
// MI455X (gfx1250) — compile-verified, matrix operand order checked by reading, not run on hardware
//
#include <hip/hip_runtime.h>
#include <hip/hip_bf16.h>
#include <math.h>

typedef __attribute__((ext_vector_type(16))) _Float16 v16h;
typedef __attribute__((ext_vector_type(8)))  _Float16 v8h;
typedef __attribute__((ext_vector_type(8)))  float    v8f;

#define NROWS 2048
#define DIN   256
#define DHID  32
#define DOUT  256
#define DCAT  512

// ---------------------------------------------------------------------------
// Attention-feature MLPs: one wave (32 threads) per row; rows [0,2048) = x,
// rows [2048,4096) = neibs.  out = tanh(row @ W1 + b1) @ W2 + b2  -> f16
// ---------------------------------------------------------------------------
__global__ void k_att(const float* __restrict__ x, const float* __restrict__ nb,
                      const float* __restrict__ w1x, const float* __restrict__ b1x,
                      const float* __restrict__ w2x, const float* __restrict__ b2x,
                      const float* __restrict__ w1n, const float* __restrict__ b1n,
                      const float* __restrict__ w2n, const float* __restrict__ b2n,
                      _Float16* __restrict__ xatt, _Float16* __restrict__ natt) {
    int row = blockIdx.x;
    int t   = threadIdx.x;            // 0..31 == hidden unit index
    const float *src, *w1, *b1, *w2, *b2;
    _Float16* dst;
    if (row < NROWS) {
        src = x + (size_t)row * DIN;
        w1 = w1x; b1 = b1x; w2 = w2x; b2 = b2x;
        dst = xatt + (size_t)row * DHID;
    } else {
        int r = row - NROWS;
        src = nb + (size_t)r * DIN;
        w1 = w1n; b1 = b1n; w2 = w2n; b2 = b2n;
        dst = natt + (size_t)r * DHID;
    }
    __shared__ float h[DHID];
    float acc = b1[t];
    #pragma unroll 4
    for (int k = 0; k < DIN; ++k) acc += src[k] * w1[k * DHID + t];
    h[t] = tanhf(acc);
    __syncthreads();
    float o = b2[t];
    #pragma unroll
    for (int j = 0; j < DHID; ++j) o += h[j] * w2[j * DHID + t];
    dst[t] = (_Float16)o;
}

// ---------------------------------------------------------------------------
// f32 (R x C, row-major)  ->  f16 transposed (C x R, row-major)
// ---------------------------------------------------------------------------
__global__ void k_cvt_t(const float* __restrict__ src, _Float16* __restrict__ dst,
                        int R, int C) {
    int i = blockIdx.x * blockDim.x + threadIdx.x;
    if (i >= R * C) return;
    int r = i / C, c = i % C;
    dst[(size_t)c * R + r] = (_Float16)src[i];
}

// pack x (f32) into the first DIN columns of xcat (f16, ld = DCAT)
__global__ void k_pack_x(const float* __restrict__ x, _Float16* __restrict__ xcat) {
    int i = blockIdx.x * blockDim.x + threadIdx.x;
    if (i >= NROWS * DIN) return;
    int r = i / DIN, c = i % DIN;
    xcat[(size_t)r * DCAT + c] = (_Float16)x[i];
}

// sigmoid(agg) -> second DIN columns of xcat (f16)
__global__ void k_sig_pack(const float* __restrict__ agg, _Float16* __restrict__ xcat) {
    int i = blockIdx.x * blockDim.x + threadIdx.x;
    if (i >= NROWS * DIN) return;
    int r = i / DIN, c = i % DIN;
    float v = 1.0f / (1.0f + __expf(-agg[i]));
    xcat[(size_t)r * DCAT + DIN + c] = (_Float16)v;
}

// out = sigmoid(tmp + bias)  (final f32 output)
__global__ void k_out(const float* __restrict__ tmp, const float* __restrict__ bias,
                      float* __restrict__ out) {
    int i = blockIdx.x * blockDim.x + threadIdx.x;
    if (i >= NROWS * DOUT) return;
    int c = i % DOUT;
    out[i] = 1.0f / (1.0f + __expf(-(tmp[i] + bias[c])));
}

// ---------------------------------------------------------------------------
// Row softmax with multiplicative mask (exactly reference semantics):
// p = softmax(S[row,:] * mask[row,:])   -> f16 probabilities
// One 256-thread block per row; 8 elements per thread.
// ---------------------------------------------------------------------------
__global__ void k_softmax(const float* __restrict__ S, const float* __restrict__ mask,
                          _Float16* __restrict__ P) {
    int row = blockIdx.x;
    int t   = threadIdx.x;                      // 0..255
    const size_t base = (size_t)row * NROWS;
    __shared__ float red[256];

    float v[8];
    float m = -3.0e38f;
    #pragma unroll
    for (int j = 0; j < 8; ++j) {
        int c = t + j * 256;
        float s = S[base + c] * mask[base + c];
        v[j] = s;
        m = fmaxf(m, s);
    }
    red[t] = m; __syncthreads();
    for (int s = 128; s > 0; s >>= 1) {
        if (t < s) red[t] = fmaxf(red[t], red[t + s]);
        __syncthreads();
    }
    m = red[0]; __syncthreads();

    float sum = 0.0f;
    #pragma unroll
    for (int j = 0; j < 8; ++j) { v[j] = __expf(v[j] - m); sum += v[j]; }
    red[t] = sum; __syncthreads();
    for (int s = 128; s > 0; s >>= 1) {
        if (t < s) red[t] += red[t + s];
        __syncthreads();
    }
    float inv = 1.0f / red[0];
    #pragma unroll
    for (int j = 0; j < 8; ++j)
        P[base + t + j * 256] = (_Float16)(v[j] * inv);
}

// ---------------------------------------------------------------------------
// WMMA GEMM, register-blocked 16x64 per wave:
//   C(MxN, f32) = A(MxK, f16 row-major, lda=K) @ B, with B supplied as
//   Bt = B^T (NxK, f16 row-major, ldb=K).
// Per K-step (32): 1 A tile + 4 B tiles (10 x b128 loads) -> 4 independent
// v_wmma_f32_16x16x32_f16 accumulator chains (good XDL pipelining, 4x A reuse).
//
// 16-bit A-tile VGPR layout (ISA 7.12.2): lane L holds major index (L&15);
// lanes 0-15 carry K = k0+[0..7] and k0+[16..23], lanes 16-31 carry
// K = k0+[8..15] and k0+[24..31].  Same loader serves A and Bt.
// ---------------------------------------------------------------------------
__device__ __forceinline__ v16h load_tile16x32(const _Float16* __restrict__ base,
                                               int ld, int major0, int k0, int lane) {
    int m  = major0 + (lane & 15);
    int kb = k0 + ((lane >> 4) << 3);
    const _Float16* p = base + (size_t)m * ld + kb;
    v8h lo = *(const v8h*)(p);        // K = kb .. kb+7
    v8h hi = *(const v8h*)(p + 16);   // K = kb+16 .. kb+23
    v16h r;
    #pragma unroll
    for (int i = 0; i < 8; ++i) { r[i] = lo[i]; r[i + 8] = hi[i]; }
    return r;
}

__device__ __forceinline__ void store_c16x16(float* __restrict__ C, int N,
                                             int m0, int n0, int lane, const v8f& c) {
    // C/D layout: lanes 0-15 -> N=n0+lane, rows m0+0..7 ; lanes 16-31 -> rows m0+8..15
    int n  = n0 + (lane & 15);
    int mb = m0 + ((lane >> 4) << 3);
    #pragma unroll
    for (int r = 0; r < 8; ++r)
        C[(size_t)(mb + r) * N + n] = c[r];
}

__global__ void k_gemm16x64(const _Float16* __restrict__ A, const _Float16* __restrict__ Bt,
                            float* __restrict__ C, int M, int N, int K) {
    int lane = threadIdx.x;           // blockDim.x == 32 (one wave)
    int n0 = blockIdx.x * 64;
    int m0 = blockIdx.y * 16;
    (void)M;

    v8f acc0 = {}, acc1 = {}, acc2 = {}, acc3 = {};
    for (int k0 = 0; k0 < K; k0 += 32) {
        v16h a  = load_tile16x32(A,  K, m0,      k0, lane);
        v16h b0 = load_tile16x32(Bt, K, n0,      k0, lane);
        v16h b1 = load_tile16x32(Bt, K, n0 + 16, k0, lane);
        v16h b2 = load_tile16x32(Bt, K, n0 + 32, k0, lane);
        v16h b3 = load_tile16x32(Bt, K, n0 + 48, k0, lane);
        acc0 = __builtin_amdgcn_wmma_f32_16x16x32_f16(false, a, false, b0,
                                                      (short)0, acc0, false, false);
        acc1 = __builtin_amdgcn_wmma_f32_16x16x32_f16(false, a, false, b1,
                                                      (short)0, acc1, false, false);
        acc2 = __builtin_amdgcn_wmma_f32_16x16x32_f16(false, a, false, b2,
                                                      (short)0, acc2, false, false);
        acc3 = __builtin_amdgcn_wmma_f32_16x16x32_f16(false, a, false, b3,
                                                      (short)0, acc3, false, false);
    }

    store_c16x16(C, N, m0, n0,      lane, acc0);
    store_c16x16(C, N, m0, n0 + 16, lane, acc1);
    store_c16x16(C, N, m0, n0 + 32, lane, acc2);
    store_c16x16(C, N, m0, n0 + 48, lane, acc3);
}

// ---------------------------------------------------------------------------
// Host-side orchestration
// ---------------------------------------------------------------------------
extern "C" void kernel_launch(void* const* d_in, const int* in_sizes, int n_in,
                              void* d_out, int out_size, void* d_ws, size_t ws_size,
                              hipStream_t stream) {
    (void)in_sizes; (void)n_in; (void)out_size; (void)ws_size;

    const float* x      = (const float*)d_in[0];
    const float* neibs  = (const float*)d_in[1];
    // d_in[2] = edge_emb : dead code in reference, intentionally unused
    const float* mask   = (const float*)d_in[3];
    const float* ax_w1  = (const float*)d_in[4];
    const float* ax_b1  = (const float*)d_in[5];
    const float* ax_w2  = (const float*)d_in[6];
    const float* ax_b2  = (const float*)d_in[7];
    const float* an_w1  = (const float*)d_in[8];
    const float* an_b1  = (const float*)d_in[9];
    const float* an_w2  = (const float*)d_in[10];
    const float* an_b2  = (const float*)d_in[11];
    // d_in[12..15] = ae_* : dead code, unused
    const float* fcx_w  = (const float*)d_in[16];
    const float* fcx_b  = (const float*)d_in[17];
    float* out = (float*)d_out;

    // ---- scratch carve-out (256B aligned) ----
    char* w = (char*)d_ws;
    auto carve = [&](size_t bytes) -> void* {
        void* p = (void*)w;
        w += (bytes + 255) & ~(size_t)255;
        return p;
    };
    _Float16* s_xatt   = (_Float16*)carve((size_t)NROWS * DHID * 2);   // 128 KB
    _Float16* s_natt   = (_Float16*)carve((size_t)NROWS * DHID * 2);   // 128 KB
    _Float16* s_neibsT = (_Float16*)carve((size_t)DIN * NROWS * 2);    // 1 MB  (256 x 2048)
    _Float16* s_fcxwT  = (_Float16*)carve((size_t)DOUT * DCAT * 2);    // 256 KB (256 x 512)
    _Float16* s_xcat   = (_Float16*)carve((size_t)NROWS * DCAT * 2);   // 2 MB
    float*    s_agg    = (float*)   carve((size_t)NROWS * DIN * 4);    // 2 MB
    float*    s_tmp    = (float*)   carve((size_t)NROWS * DOUT * 4);   // 2 MB
    float*    s_S      = (float*)   carve((size_t)NROWS * NROWS * 4);  // 16 MB
    _Float16* s_P      = (_Float16*)carve((size_t)NROWS * NROWS * 2);  // 8 MB

    // 1) attention features for x and neibs (f16)
    k_att<<<2 * NROWS, 32, 0, stream>>>(x, neibs,
                                        ax_w1, ax_b1, ax_w2, ax_b2,
                                        an_w1, an_b1, an_w2, an_b2,
                                        s_xatt, s_natt);

    // 2) operand prep: neibs^T (f16), fcx_w^T (f16), xcat[:, :256] = f16(x)
    {
        int n1 = NROWS * DIN;
        k_cvt_t<<<(n1 + 255) / 256, 256, 0, stream>>>(neibs, s_neibsT, NROWS, DIN);
        int n2 = DCAT * DOUT;
        k_cvt_t<<<(n2 + 255) / 256, 256, 0, stream>>>(fcx_w, s_fcxwT, DCAT, DOUT);
        k_pack_x<<<(n1 + 255) / 256, 256, 0, stream>>>(x, s_xcat);
    }

    // 3) raw scores: S = x_att @ neib_att^T   (2048 x 2048, K=32)
    k_gemm16x64<<<dim3(NROWS / 64, NROWS / 16), 32, 0, stream>>>(
        s_xatt, s_natt, s_S, NROWS, NROWS, DHID);

    // 4) masked softmax -> f16 probabilities
    k_softmax<<<NROWS, 256, 0, stream>>>(s_S, mask, s_P);

    // 5) aggregation: agg = P @ neibs   (2048 x 256, K=2048) — dominant GEMM
    k_gemm16x64<<<dim3(DIN / 64, NROWS / 16), 32, 0, stream>>>(
        s_P, s_neibsT, s_agg, NROWS, DIN, NROWS);

    // 6) sigmoid(agg) into xcat[:, 256:512]
    k_sig_pack<<<(NROWS * DIN + 255) / 256, 256, 0, stream>>>(s_agg, s_xcat);

    // 7) final FC: tmp = xcat @ fcx_w   (2048 x 256, K=512)
    k_gemm16x64<<<dim3(DOUT / 64, NROWS / 16), 32, 0, stream>>>(
        s_xcat, s_fcxwT, s_tmp, NROWS, DOUT, DCAT);

    // 8) out = sigmoid(tmp + fcx_b)
    k_out<<<(NROWS * DOUT + 255) / 256, 256, 0, stream>>>(s_tmp, fcx_b, out);
}
